// RoEmbedding_2929167696287
// MI455X (gfx1250) — matrix operand order repaired and checked by reading, not compiled, not run on hardware
//
#include <hip/hip_runtime.h>

// -------------------------------------------------------------------------
// Rope-pair logits + weighted NLL for MI455X (gfx1250, wave32, WMMA).
//
// Pipeline:
//   k0: one-pass fp32 -> split-bf16 (hi/lo) conversion of X and W, so the
//       GEMM hot loops issue zero conversion VALU and stream half the bytes.
//   k1: GEMM1 (X @ W^T + b) with fused RoPE epilogue; writes q/k as
//       split-bf16 to workspace. 3-product hi/lo bf16 WMMA (~fp32 accuracy).
//   k2: GEMM2 per (batch, head): Q @ K^T -> logits (B,S,S,2), both heads
//       per wave so the h-interleaved store is a single pass.
//   k3/k4: deterministic two-stage weighted-NLL reduction -> d_out[0].
// -------------------------------------------------------------------------

#define BB   8
#define SS   1024
#define CIN  1024
#define DQKc 256
#define NOUT 1024                    // 4*DQK
#define PAIR_POS_W 1.0f
#define LN_ROPE 2.70805020110221007f // ln(15)

typedef __attribute__((ext_vector_type(16))) __bf16 v16bf;
typedef __attribute__((ext_vector_type(8)))  __bf16 v8bf;
typedef __attribute__((ext_vector_type(4)))  __bf16 v4bf;
typedef __attribute__((ext_vector_type(8)))  float  v8f;
typedef __attribute__((ext_vector_type(4)))  float  v4f;

union V16U { v16bf v; v8bf h[2]; };

static __device__ inline v8f wmma_bf16(v16bf a, v16bf b, v8f c) {
  return __builtin_amdgcn_wmma_f32_16x16x32_bf16(false, a, false, b,
                                                 (short)0, c, false, false);
}

// ---------------------------------------------------------------------
// Kernel 0: x -> (hi = bf16(x), lo = bf16(x - hi)). Pure streaming.
// ---------------------------------------------------------------------
__global__ __launch_bounds__(256) void split_bf16_kernel(
    const float* __restrict__ src, __bf16* __restrict__ hi,
    __bf16* __restrict__ lo, unsigned int nquads)
{
  unsigned int q = blockIdx.x * blockDim.x + threadIdx.x;
  if (q >= nquads) return;
  v4f x = *(const v4f*)(src + (size_t)q * 4);
  v4bf h, l;
#pragma unroll
  for (int j = 0; j < 4; ++j) {
    __bf16 hv = (__bf16)x[j];
    h[j] = hv;
    l[j] = (__bf16)(x[j] - (float)hv);
  }
  *(v4bf*)(hi + (size_t)q * 4) = h;
  *(v4bf*)(lo + (size_t)q * 4) = l;
}

// ---------------------------------------------------------------------
// Kernel 1: out = X @ W^T + b, fused RoPE, store split-bf16 q/k.
// One wave per 32x32 output tile (2x2 WMMA register blocking).
// A fragment (16x32 bf16): lane m = lane&15, K(j) = j + 16*(lane>>4)
// B fragment (32x16 bf16): lane n = lane&15, K(j) = j + 16*(lane>>4)
// C fragment (16x16 f32) : VGPR r -> m = r + 8*(lane>>4), n = lane&15
// ---------------------------------------------------------------------
__global__ __launch_bounds__(32) void gemm1_rope_kernel(
    const __bf16* __restrict__ Xhi, const __bf16* __restrict__ Xlo,
    const __bf16* __restrict__ Whi, const __bf16* __restrict__ Wlo,
    const float* __restrict__ bias,
    __bf16* __restrict__ qhi, __bf16* __restrict__ qlo,
    __bf16* __restrict__ khi, __bf16* __restrict__ klo)
{
  const int lane  = threadIdx.x;
  const int hl    = lane >> 4;
  const int ln    = lane & 15;
  const int tileM = blockIdx.x * 32;
  const int tileN = blockIdx.y * 32;

  v8f acc[2][2] = {};

  for (int kk = 0; kk < CIN; kk += 32) {
    V16U ahi[2], alo[2];
    v16bf bhi[2], blo[2];
#pragma unroll
    for (int t = 0; t < 2; ++t) {
      const int m = tileM + 16 * t + ln;
      const size_t ro = (size_t)m * CIN + kk + 16 * hl;
      ahi[t].h[0] = *(const v8bf*)(Xhi + ro);
      ahi[t].h[1] = *(const v8bf*)(Xhi + ro + 8);
      alo[t].h[0] = *(const v8bf*)(Xlo + ro);
      alo[t].h[1] = *(const v8bf*)(Xlo + ro + 8);
      const int n = tileN + 16 * t + ln;
      const size_t co = (size_t)n * CIN + kk + 16 * hl;
      bhi[t] = *(const v16bf*)(Whi + co);
      blo[t] = *(const v16bf*)(Wlo + co);
    }
#pragma unroll
    for (int tm = 0; tm < 2; ++tm)
#pragma unroll
      for (int tn = 0; tn < 2; ++tn) {
        acc[tm][tn] = wmma_bf16(ahi[tm].v, bhi[tn], acc[tm][tn]);
        acc[tm][tn] = wmma_bf16(ahi[tm].v, blo[tn], acc[tm][tn]);
        acc[tm][tn] = wmma_bf16(alo[tm].v, bhi[tn], acc[tm][tn]);
      }
  }

  float bv[2];
#pragma unroll
  for (int t = 0; t < 2; ++t) bv[t] = bias[tileN + 16 * t + ln];

  // Fused bias + RoPE epilogue. Pair partner (n^1) sits in lane^1.
#pragma unroll
  for (int tm = 0; tm < 2; ++tm) {
#pragma unroll
    for (int tn = 0; tn < 2; ++tn) {
      const int n  = tileN + 16 * tn + ln;
      const int dd = n & (DQKc - 1);
      const int f  = dd >> 1;
      const float invf = expf(-(float)f * (LN_ROPE / 128.0f)); // 15^(-f/128)
#pragma unroll
      for (int r = 0; r < 8; ++r) {
        const int m = tileM + 16 * tm + r + 8 * hl;
        const int s = m & (SS - 1);
        float v = acc[tm][tn][r] + bv[tn];
        float p = __shfl_xor(v, 1, 32);
        float ang = (float)s * invf;
        float cs = cosf(ang), sn = sinf(ang);
        float rot = ((lane & 1) == 0) ? (v * cs - p * sn)   // even d: x*c - x[d+1]*s
                                      : (v * cs + p * sn);  // odd  d: x*c + x[d-1]*s
        const int hID  = n >> 9;         // head
        const int isk  = (n >> 8) & 1;   // q or k half
        const int bidx = m >> 10;        // batch
        const size_t dst = (((size_t)(bidx * 2 + hID)) * SS + s) * DQKc + dd;
        __bf16 hi = (__bf16)rot;
        __bf16 lo = (__bf16)(rot - (float)hi);
        if (isk) { khi[dst] = hi; klo[dst] = lo; }
        else     { qhi[dst] = hi; qlo[dst] = lo; }
      }
    }
  }
}

// ---------------------------------------------------------------------
// Kernel 2: logits[b,m,n,h] = qr[b,h,m,:] . kr[b,h,n,:]  (d = 256)
// One wave per (b, 32x32 mn-tile), both heads, 2x2 register blocking.
// ---------------------------------------------------------------------
__global__ __launch_bounds__(32) void gemm2_logits_kernel(
    const __bf16* __restrict__ qhi, const __bf16* __restrict__ qlo,
    const __bf16* __restrict__ khi, const __bf16* __restrict__ klo,
    float* __restrict__ logits)
{
  const int lane  = threadIdx.x;
  const int hl    = lane >> 4;
  const int ln    = lane & 15;
  const int tileM = blockIdx.x * 32;
  const int tileN = blockIdx.y * 32;
  const int b     = blockIdx.z;

  v8f acc[2][2][2] = {};   // [head][tm][tn]

#pragma unroll
  for (int h = 0; h < 2; ++h) {
    const size_t base = (size_t)(b * 2 + h) * SS * DQKc;
    const __bf16* qh = qhi + base;  const __bf16* ql = qlo + base;
    const __bf16* kh = khi + base;  const __bf16* kl = klo + base;
#pragma unroll
    for (int kk = 0; kk < DQKc; kk += 32) {
      V16U ahi[2], alo[2];
      v16bf bhi[2], blo[2];
#pragma unroll
      for (int t = 0; t < 2; ++t) {
        const int m = tileM + 16 * t + ln;
        const size_t ro = (size_t)m * DQKc + kk + 16 * hl;
        ahi[t].h[0] = *(const v8bf*)(qh + ro);
        ahi[t].h[1] = *(const v8bf*)(qh + ro + 8);
        alo[t].h[0] = *(const v8bf*)(ql + ro);
        alo[t].h[1] = *(const v8bf*)(ql + ro + 8);
        const int n = tileN + 16 * t + ln;
        const size_t co = (size_t)n * DQKc + kk + 16 * hl;
        bhi[t] = *(const v16bf*)(kh + co);
        blo[t] = *(const v16bf*)(kl + co);
      }
#pragma unroll
      for (int tm = 0; tm < 2; ++tm)
#pragma unroll
        for (int tn = 0; tn < 2; ++tn) {
          acc[h][tm][tn] = wmma_bf16(ahi[tm].v, bhi[tn], acc[h][tm][tn]);
          acc[h][tm][tn] = wmma_bf16(ahi[tm].v, blo[tn], acc[h][tm][tn]);
          acc[h][tm][tn] = wmma_bf16(alo[tm].v, bhi[tn], acc[h][tm][tn]);
        }
    }
  }

#pragma unroll
  for (int tm = 0; tm < 2; ++tm)
#pragma unroll
    for (int tn = 0; tn < 2; ++tn) {
      const int n = tileN + 16 * tn + ln;
#pragma unroll
      for (int r = 0; r < 8; ++r) {
        const int m = tileM + 16 * tm + r + 8 * hl;
        const size_t o = ((((size_t)b * SS + m) * SS) + n) * 2;
        logits[o]     = acc[0][tm][tn][r];
        logits[o + 1] = acc[1][tm][tn][r];
      }
    }
}

// ---------------------------------------------------------------------
// Kernel 3/4: deterministic weighted-NLL reduction (no float atomics).
// ---------------------------------------------------------------------
__global__ void loss_partial_kernel(const float* __restrict__ logits,
                                    const int* __restrict__ labels,
                                    const int* __restrict__ masks,
                                    float2* __restrict__ partials)
{
  __shared__ float sA[256], sB[256];
  const int tid = threadIdx.x;
  const long long nflat  = (long long)BB * SS * SS;
  const long long stride = (long long)gridDim.x * blockDim.x;
  float a = 0.0f, wsum = 0.0f;
  for (long long i = (long long)blockIdx.x * blockDim.x + tid; i < nflat; i += stride) {
    float z0 = logits[2 * i];
    float z1 = logits[2 * i + 1];
    int lab = labels[i];
    float w = (masks[i] == 1) ? ((lab == 1) ? PAIR_POS_W : 1.0f) : 0.0f;
    float mx  = fmaxf(z0, z1);
    float lse = mx + logf(expf(z0 - mx) + expf(z1 - mx));
    float nll = lse - (lab ? z1 : z0);
    a    += w * nll;
    wsum += w;
  }
  sA[tid] = a; sB[tid] = wsum;
  __syncthreads();
  for (int o = 128; o > 0; o >>= 1) {
    if (tid < o) { sA[tid] += sA[tid + o]; sB[tid] += sB[tid + o]; }
    __syncthreads();
  }
  if (tid == 0) partials[blockIdx.x] = make_float2(sA[0], sB[0]);
}

__global__ void loss_final_kernel(const float2* __restrict__ partials, int n,
                                  float* __restrict__ out0)
{
  __shared__ float sA[256], sB[256];
  const int tid = threadIdx.x;
  float a = 0.0f, b = 0.0f;
  for (int i = tid; i < n; i += 256) { float2 p = partials[i]; a += p.x; b += p.y; }
  sA[tid] = a; sB[tid] = b;
  __syncthreads();
  for (int o = 128; o > 0; o >>= 1) {
    if (tid < o) { sA[tid] += sA[tid + o]; sB[tid] += sB[tid + o]; }
    __syncthreads();
  }
  if (tid == 0) out0[0] = sA[0] / sB[0];
}

// ---------------------------------------------------------------------
extern "C" void kernel_launch(void* const* d_in, const int* in_sizes, int n_in,
                              void* d_out, int out_size, void* d_ws, size_t ws_size,
                              hipStream_t stream)
{
  (void)in_sizes; (void)n_in; (void)out_size; (void)ws_size;

  const float* X      = (const float*)d_in[0];  // (B,S,IN)
  const int*   labels = (const int*)  d_in[1];  // (B,S,S)
  const int*   masks  = (const int*)  d_in[2];  // (B,S,S)
  const float* W      = (const float*)d_in[3];  // (4*DQK, IN)
  const float* bias   = (const float*)d_in[4];  // (4*DQK,)
  float* out = (float*)d_out;                   // [loss(1), logits(B*S*S*2)]

  // Workspace layout (bytes):
  //   [0,   32M): q/k split-bf16 (4 arrays x 8 MiB)
  //   [32M, 64M): X split-bf16 (hi 16 MiB, lo 16 MiB)
  //   [64M, 68M): W split-bf16 (hi 2 MiB, lo 2 MiB)
  //   [68M, ...): loss partials (1024 float2)
  const size_t QKN = (size_t)BB * 2 * SS * DQKc;   // 4,194,304
  const size_t XN  = (size_t)BB * SS * CIN;        // 8,388,608
  const size_t WN  = (size_t)NOUT * CIN;           // 1,048,576
  __bf16* qhi = (__bf16*)d_ws;
  __bf16* qlo = qhi + QKN;
  __bf16* khi = qlo + QKN;
  __bf16* klo = khi + QKN;
  __bf16* Xhi = klo + QKN;
  __bf16* Xlo = Xhi + XN;
  __bf16* Whi = Xlo + XN;
  __bf16* Wlo = Whi + WN;
  float2* partials = (float2*)(Wlo + WN);

  const unsigned int xq = (unsigned int)(XN / 4);
  const unsigned int wq = (unsigned int)(WN / 4);
  split_bf16_kernel<<<(xq + 255) / 256, 256, 0, stream>>>(X, Xhi, Xlo, xq);
  split_bf16_kernel<<<(wq + 255) / 256, 256, 0, stream>>>(W, Whi, Wlo, wq);

  gemm1_rope_kernel<<<dim3((BB * SS) / 32, NOUT / 32), 32, 0, stream>>>(
      Xhi, Xlo, Whi, Wlo, bias, qhi, qlo, khi, klo);
  gemm2_logits_kernel<<<dim3(SS / 32, SS / 32, BB), 32, 0, stream>>>(
      qhi, qlo, khi, klo, out + 1);
  loss_partial_kernel<<<1024, 256, 0, stream>>>(out + 1, labels, masks, partials);
  loss_final_kernel<<<1, 256, 0, stream>>>(partials, 1024, out);
}
